// RSNN_88064009437998
// MI455X (gfx1250) — matrix operand order repaired and checked by reading, not compiled, not run on hardware
//
#include <hip/hip_runtime.h>
#include <hip/hip_bf16.h>

// ---------------------------------------------------------------------------
// RSNN scan for MI455X (gfx1250, wave32):
//   GEMM1: I  = Xf16 [8192x256]  @ W_in^T  (+bias) -> f32
//   SCAN : per-(b,n) 256-step LIF recurrence (elementwise)
//   GEMM2: Z2 = Zf16 [8192x2048] @ W_bern^T        -> f32
//   GATHER: z2s[t,b,j] = Z2[t*32+b, idx[t,j]]
// GEMMs: 128x128 block tile, double-buffered LDS staging via
// global_load_async_to_lds_b128 (ASYNCcnt), 32x64 wave tile, 8 WMMA/k-step.
// ---------------------------------------------------------------------------

typedef __attribute__((ext_vector_type(16))) _Float16 v16h;
typedef __attribute__((ext_vector_type(8)))  _Float16 v8h;
typedef __attribute__((ext_vector_type(8)))  float    v8f;

#define T_STEPS 256
#define BATCH   32
#define N_IN    256
#define N_NEUR  2048
#define DECAY_F 0.9048374180359595f   // exp(-1/10)

#define BM 128
#define BN 128
#define BK 32

// ---- WMMA fragment loads from LDS (layouts per CDNA5 ISA §7.12.2) ----------
// A (16x32 f16): row m = lane%16; lanes 0-15 hold K={0..7,16..23},
// lanes 16-31 hold K={8..15,24..31} -> two contiguous 16B ds loads.
__device__ __forceinline__ v16h lds_a_frag(const _Float16* s, int m, int lane) {
    const int koff = (lane >> 4) * 8;
    const _Float16* p = s + m * BK + koff;
    v8h lo = *(const v8h*)(p);
    v8h hi = *(const v8h*)(p + 16);
    v16h r;
#pragma unroll
    for (int i = 0; i < 8; ++i) { r[i] = lo[i]; r[i + 8] = hi[i]; }
    return r;
}

// B (32x16 f16) from transposed tile [n x k]: col n = lane%16; lanes 0-15 hold
// K={0..7,16..23}, lanes 16-31 hold K={8..15,24..31} -> two 16B ds loads.
__device__ __forceinline__ v16h lds_b_frag(const _Float16* s, int n, int lane) {
    const int kb = (lane >> 4) * 8;
    const _Float16* p = s + n * BK + kb;
    v8h lo = *(const v8h*)(p);
    v8h hi = *(const v8h*)(p + 16);
    v16h r;
#pragma unroll
    for (int i = 0; i < 8; ++i) { r[i] = lo[i]; r[i + 8] = hi[i]; }
    return r;
}

// ---- Async stage: global -> LDS, 16KB per block k-step ---------------------
// Each of 256 threads issues 4x global_load_async_to_lds_b128 (ASYNCcnt).
__device__ __forceinline__ void async_stage(const _Float16* __restrict__ gA,
                                            const _Float16* __restrict__ gB,
                                            _Float16* sA, _Float16* sB,
                                            int K, int blockM, int blockN,
                                            int k0, int tid) {
#pragma unroll
    for (int i = 0; i < 2; ++i) {
        const int c   = tid + i * 256;     // chunk 0..511 (128 rows x 4 chunks)
        const int row = c >> 2;
        const int off = (c & 3) * 8;       // in halves (16B chunks)
        unsigned long long ga =
            (unsigned long long)(gA + (size_t)(blockM + row) * K + k0 + off);
        unsigned la = (unsigned)(uintptr_t)(sA + row * BK + off);
        asm volatile("global_load_async_to_lds_b128 %0, %1, off"
                     :: "v"(la), "v"(ga) : "memory");
        unsigned long long gb =
            (unsigned long long)(gB + (size_t)(blockN + row) * K + k0 + off);
        unsigned lb = (unsigned)(uintptr_t)(sB + row * BK + off);
        asm volatile("global_load_async_to_lds_b128 %0, %1, off"
                     :: "v"(lb), "v"(gb) : "memory");
    }
}

// ---- NT WMMA GEMM: C[MxN] = A[MxK] * BT[NxK]^T (+ optional bias[N]) --------
// 256 threads = 8 waves (4 in M x 2 in N); wave tile 32x64 (2x4 fragments).
__global__ __launch_bounds__(256)
void wmma_gemm_nt(const _Float16* __restrict__ A, const _Float16* __restrict__ BT,
                  float* __restrict__ C, const float* __restrict__ bias,
                  int M, int N, int K) {
    __shared__ __align__(16) _Float16 As[2][BM * BK];   // 2 x 8 KB
    __shared__ __align__(16) _Float16 Bs[2][BN * BK];   // 2 x 8 KB

    const int tid   = threadIdx.x;
    const int lane  = tid & 31;
    const int wave  = tid >> 5;
    const int wm    = (wave & 3) * 32;      // wave M offset in block tile
    const int wn    = (wave >> 2) * 64;     // wave N offset in block tile
    const int row   = lane & 15;
    const int blockM = blockIdx.y * BM;
    const int blockN = blockIdx.x * BN;
    const int KT = K >> 5;

    v8f acc[2][4] = {};

    async_stage(A, BT, &As[0][0], &Bs[0][0], K, blockM, blockN, 0, tid);

    for (int kt = 0; kt < KT; ++kt) {
        const int buf = kt & 1;
        if (kt + 1 < KT) {
            async_stage(A, BT, &As[buf ^ 1][0], &Bs[buf ^ 1][0],
                        K, blockM, blockN, (kt + 1) << 5, tid);
            asm volatile("s_wait_asynccnt 0x4" ::: "memory"); // stage kt resident
        } else {
            asm volatile("s_wait_asynccnt 0x0" ::: "memory");
        }
        __syncthreads();                    // publish LDS across all 8 waves

        v16h af[2];
#pragma unroll
        for (int i = 0; i < 2; ++i)
            af[i] = lds_a_frag(&As[buf][0], wm + i * 16 + row, lane);
        v16h bf[4];
#pragma unroll
        for (int j = 0; j < 4; ++j)
            bf[j] = lds_b_frag(&Bs[buf][0], wn + j * 16 + row, lane);
#pragma unroll
        for (int i = 0; i < 2; ++i)
#pragma unroll
            for (int j = 0; j < 4; ++j)
                acc[i][j] = __builtin_amdgcn_wmma_f32_16x16x32_f16(
                    false, af[i], false, bf[j], (short)0, acc[i][j], false, false);

        __syncthreads();                    // buffer reusable for stage kt+2
    }

    // C layout: VGPR r -> M = r + (lane>=16 ? 8 : 0); N = lane%16.
    const int rHi = (lane >> 4) * 8;
#pragma unroll
    for (int i = 0; i < 2; ++i) {
#pragma unroll
        for (int j = 0; j < 4; ++j) {
            const int n = blockN + wn + j * 16 + row;
            const float bj = bias ? bias[n] : 0.0f;
#pragma unroll
            for (int r = 0; r < 8; ++r) {
                const int m = blockM + wm + i * 16 + rHi + r;
                C[(size_t)m * N + n] = acc[i][j][r] + bj;
            }
        }
    }
}

// ---- Elementwise LIF scan over T; one thread per (b,n) ---------------------
__global__ __launch_bounds__(256)
void scan_kernel(const float* __restrict__ I, float* __restrict__ zs,
                 float* __restrict__ vs, _Float16* __restrict__ Z) {
    const int gid = blockIdx.x * 256 + threadIdx.x;   // 0 .. B*N-1
    const int n = gid & (N_NEUR - 1);
    const int b = gid >> 11;
    float v = 0.0f;
#pragma unroll 4
    for (int t = 0; t < T_STEPS; ++t) {
        const size_t r = (size_t)(t * BATCH + b) * N_NEUR + n;
        float nv = fminf(fmaxf(DECAY_F * v + I[r], -1.0f), 1.01f);
        const float z = (nv > 1.0f) ? 1.0f : 0.0f;
        if (z > 0.0f) nv = -1.0f;                      // hard reset
        zs[r] = z;
        vs[r] = nv;
        Z[r]  = (_Float16)z;
        v = nv;
    }
}

// ---- Column gather epilogue: z2s[t,b,j] = Z2[t*32+b, idx[t,j]] -------------
__global__ __launch_bounds__(256)
void gather_kernel(const float* __restrict__ Z2, const int* __restrict__ idx,
                   float* __restrict__ z2s) {
    const int gid = blockIdx.x * 256 + threadIdx.x;   // 0 .. T*B*N-1
    const int n  = gid & (N_NEUR - 1);
    const int tb = gid >> 11;
    const int t  = tb >> 5;
    const int col = idx[t * N_NEUR + n];
    z2s[gid] = Z2[(size_t)tb * N_NEUR + col];
}

// ---- Layout / precision prep ----------------------------------------------
__global__ __launch_bounds__(256)
void cvt_f16_kernel(const float* __restrict__ in, _Float16* __restrict__ out, int n) {
    const int gid = blockIdx.x * 256 + threadIdx.x;
    if (gid < n) out[gid] = (_Float16)in[gid];
}

// in: row-major [K x N] f32  ->  out: row-major [N x K] f16 (i.e. B^T)
__global__ __launch_bounds__(256)
void transpose_cvt_kernel(const float* __restrict__ in, _Float16* __restrict__ out,
                          int K, int N) {
    const int gid = blockIdx.x * 256 + threadIdx.x;
    if (gid >= K * N) return;
    const int n = gid / K;
    const int k = gid - n * K;
    out[gid] = (_Float16)in[(size_t)k * N + n];
}

// ---------------------------------------------------------------------------
extern "C" void kernel_launch(void* const* d_in, const int* in_sizes, int n_in,
                              void* d_out, int out_size, void* d_ws, size_t ws_size,
                              hipStream_t stream) {
    const float* x      = (const float*)d_in[0];   // [T, B, N_IN]
    const float* W_in   = (const float*)d_in[1];   // [N_IN, N]
    const float* bvec   = (const float*)d_in[2];   // [N]
    const float* W_bern = (const float*)d_in[3];   // [N, N]
    const int*   idx    = (const int*)  d_in[4];   // [T, N]

    float* out = (float*)d_out;                    // [3, T, B, N]
    const size_t TBN = (size_t)T_STEPS * BATCH * N_NEUR;   // 16,777,216
    float* zs  = out;
    float* vs  = out + TBN;
    float* z2s = out + 2 * TBN;

    // Workspace carve-up (~109 MiB total):
    char* ws = (char*)d_ws;
    _Float16* Xh   = (_Float16*)ws; ws += (size_t)T_STEPS * BATCH * N_IN * 2; //  4 MiB
    _Float16* WinT = (_Float16*)ws; ws += (size_t)N_NEUR * N_IN * 2;          //  1 MiB
    _Float16* WbT  = (_Float16*)ws; ws += (size_t)N_NEUR * N_NEUR * 2;        //  8 MiB
    float*    Ibuf = (float*)ws;    ws += TBN * 4;                            // 64 MiB (reused as Z2)
    _Float16* Zh   = (_Float16*)ws;                                           // 32 MiB

    const int MROWS = T_STEPS * BATCH;             // 8192

    // 1) precision/layout prep
    cvt_f16_kernel<<<(MROWS * N_IN) / 256, 256, 0, stream>>>(x, Xh, MROWS * N_IN);
    transpose_cvt_kernel<<<(N_IN * N_NEUR) / 256, 256, 0, stream>>>(W_in, WinT, N_IN, N_NEUR);
    transpose_cvt_kernel<<<(N_NEUR * N_NEUR) / 256, 256, 0, stream>>>(W_bern, WbT, N_NEUR, N_NEUR);

    // 2) GEMM1: I = X @ W_in + b   [8192 x 2048], K = 256
    dim3 gemm_grid(N_NEUR / BN, MROWS / BM);
    wmma_gemm_nt<<<gemm_grid, 256, 0, stream>>>(Xh, WinT, Ibuf, bvec,
                                                MROWS, N_NEUR, N_IN);

    // 3) LIF scan: zs, vs outputs + f16 spike matrix Z
    scan_kernel<<<(BATCH * N_NEUR) / 256, 256, 0, stream>>>(Ibuf, zs, vs, Zh);

    // 4) GEMM2: Z2 = Z @ W_bern   [8192 x 2048], K = 2048 (reuse Ibuf)
    wmma_gemm_nt<<<gemm_grid, 256, 0, stream>>>(Zh, WbT, Ibuf, nullptr,
                                                MROWS, N_NEUR, N_NEUR);

    // 5) per-step column gather into z2s
    gather_kernel<<<(int)(TBN / 256), 256, 0, stream>>>(Ibuf, idx, z2s);
}
